// RECURSIVE_AUTOENCODER_15324443312624
// MI455X (gfx1250) — compile-verified, matrix operand order checked by reading, not run on hardware
//
#include <hip/hip_runtime.h>
#include <hip/hip_bf16.h>
#include <math.h>

// ---------------------------------------------------------------------------
// Recursive autoencoder for MI455X (gfx1250, wave32).
// One workgroup (256 thr = 8 waves) per batch column; whole 31-step recursion
// runs in LDS. GEMMs use v_wmma_f32_16x16x32_bf16. Weights are pre-swizzled
// into B-fragment order in global (L2-resident); activations keep a padded
// bf16 mirror in LDS so A fragments are two ds_load_b128, no conversion VALU.
// Per-step body specialized on M-tile count; K-loops unrolled x2 only so B
// fragments stream from L2 instead of spilling to scratch.
// ---------------------------------------------------------------------------

typedef __attribute__((ext_vector_type(16))) __bf16          v16bf;
typedef __attribute__((ext_vector_type(16))) unsigned short  v16u16;
typedef __attribute__((ext_vector_type(8)))  unsigned short  v8u16;
typedef __attribute__((ext_vector_type(4)))  unsigned        v4u32;
typedef __attribute__((ext_vector_type(8)))  float           v8f;

#define LL 32
#define BB 256
#define DD 256
#define BFP 264            // padded bf16 row stride (528B = 132 dwords -> bank skew 4)
#define EPSF 1e-9f

// fp32 -> bf16 bits, round-to-nearest-even
__device__ __forceinline__ unsigned short f2bf(float f) {
  unsigned u = __builtin_bit_cast(unsigned, f);
  u += 0x7FFFu + ((u >> 16) & 1u);
  return (unsigned short)(u >> 16);
}
__device__ __forceinline__ unsigned pack_bf16(float a, float b) {
  return (unsigned)f2bf(a) | ((unsigned)f2bf(b) << 16);
}

// K offset of VGPR v for 16-bit A/B fragments:
// lanes 0-15: V0..V3 -> K={0,2,4,6}, V4..V7 -> K={16,18,20,22}; lanes 16-31: +8
__device__ __forceinline__ int koff(int v, int hi) {
  return ((v < 4) ? (2 * v) : (16 + 2 * (v - 4))) + (hi ? 8 : 0);
}

__device__ __forceinline__ v8f wmma_bf16(v16bf a, v16bf b, v8f c) {
  return __builtin_amdgcn_wmma_f32_16x16x32_bf16(false, a, false, b,
                                                 (short)0, c, false, false);
}

// Pre-swizzled B fragment: 512 halves per (kt,nt) tile, lane-contiguous 32B.
__device__ __forceinline__ v16bf load_b_frag(const unsigned short* __restrict__ wt,
                                             int fragIdx, int lane) {
  const v16u16* p =
      reinterpret_cast<const v16u16*>(wt + (size_t)fragIdx * 512 + lane * 16);
  return __builtin_bit_cast(v16bf, *p);
}

__device__ __forceinline__ v16bf combine16(v8u16 lo, v8u16 hi) {
  v16u16 u = __builtin_shufflevector(lo, hi, 0, 1, 2, 3, 4, 5, 6, 7,
                                     8, 9, 10, 11, 12, 13, 14, 15);
  return __builtin_bit_cast(v16bf, u);
}

// A fragment from bf16 mirror of concat(x[m], x[m+1])  (K = 512 total).
// Lane's 16 halves = two contiguous 16B chunks -> 2x ds_load_b128.
__device__ __forceinline__ v16bf load_a_cat16(const unsigned short* __restrict__ xbh,
                                              int mt, int kt, int lane) {
  const int hi = lane >> 4;
  int m = mt * 16 + (lane & 15);
  int mr = (m > 30) ? 30 : m;     // pad rows read valid memory, outputs ignored
  const int k1 = kt * 32 + hi * 8;
  const int k2 = k1 + 16;         // chunks never straddle the 256 boundary
  const unsigned short* p1 =
      (k1 < DD) ? (xbh + mr * BFP + k1) : (xbh + (mr + 1) * BFP + (k1 - DD));
  const unsigned short* p2 =
      (k2 < DD) ? (xbh + mr * BFP + k2) : (xbh + (mr + 1) * BFP + (k2 - DD));
  return combine16(*(const v8u16*)p1, *(const v8u16*)p2);
}

// A fragment from bf16 mirror of plain rows (K = 256 total).
__device__ __forceinline__ v16bf load_a_row16(const unsigned short* __restrict__ src,
                                              int mt, int kt, int lane) {
  const int hi = lane >> 4;
  const int m = mt * 16 + (lane & 15);
  const unsigned short* p1 = src + m * BFP + kt * 32 + hi * 8;
  return combine16(*(const v8u16*)p1, *(const v8u16*)(p1 + 16));
}

__device__ __forceinline__ float wave_sum(float x) {
#pragma unroll
  for (int o = 16; o; o >>= 1) x += __shfl_xor(x, o, 32);
  return x;
}
__device__ __forceinline__ float wave_max(float x) {
#pragma unroll
  for (int o = 16; o; o >>= 1) x = fmaxf(x, __shfl_xor(x, o, 32));
  return x;
}

// Row softmax over rows [w*4, w*4+4) below cap; lane owns 8 contiguous cols.
// Optionally writes a packed-bf16 mirror (one ds_store_b128 per row per lane).
__device__ __forceinline__ void softmax4(float (*__restrict__ buf)[DD],
                                         unsigned short* __restrict__ mirror,
                                         int cap, int w, int lane) {
  const int c0 = lane * 8;
#pragma unroll
  for (int rr = 0; rr < 4; ++rr) {
    const int r = w * 4 + rr;          // uniform per wave
    if (r >= cap) continue;
    float v[8];
    float mx = -3.402823466e38f;
#pragma unroll
    for (int j = 0; j < 8; ++j) { v[j] = buf[r][c0 + j]; mx = fmaxf(mx, v[j]); }
    mx = wave_max(mx);
    float s = 0.f;
#pragma unroll
    for (int j = 0; j < 8; ++j) { v[j] = __expf(v[j] - mx); s += v[j]; }
    s = wave_sum(s);
    const float inv = 1.0f / s;
#pragma unroll
    for (int j = 0; j < 8; ++j) { v[j] *= inv; buf[r][c0 + j] = v[j]; }
    if (mirror) {
      v4u32 pk;
#pragma unroll
      for (int j = 0; j < 4; ++j) pk[j] = pack_bf16(v[2 * j], v[2 * j + 1]);
      *(v4u32*)(mirror + r * BFP + c0) = pk;
    }
  }
}

// KL(p || q) with reference's +EPS and renormalization of both.
__device__ __forceinline__ float kl_row(const float* __restrict__ p,
                                        const float* __restrict__ q, int lane) {
  const int c0 = lane * 8;
  float pe[8], qe[8];
  float sp = 0.f, sq = 0.f;
#pragma unroll
  for (int j = 0; j < 8; ++j) {
    pe[j] = p[c0 + j] + EPSF;
    qe[j] = q[c0 + j] + EPSF;
    sp += pe[j]; sq += qe[j];
  }
  sp = wave_sum(sp);
  sq = wave_sum(sq);
  const float c = sq / sp;   // log((p/sp)/(q/sq)) = log(p*c/q)
  float t = 0.f;
#pragma unroll
  for (int j = 0; j < 8; ++j) t += (pe[j] / sp) * __logf(pe[j] * c / qe[j]);
  return wave_sum(t);
}

// Decoder GEMM: D = hidden(bf16 mirror) @ W^T + bias  (K=256), raw -> dst.
template <int MTS>
__device__ __forceinline__ void gemm_dec(const unsigned short* __restrict__ hsh,
                                         const unsigned short* __restrict__ wt,
                                         const float* __restrict__ bias,
                                         float (*__restrict__ dst)[DD],
                                         int w, int lane) {
  v8f acc[2][MTS];
#pragma unroll
  for (int i = 0; i < 2; ++i)
#pragma unroll
    for (int j = 0; j < MTS; ++j) acc[i][j] = (v8f)0.0f;

#pragma unroll 2
  for (int kt = 0; kt < 8; ++kt) {
    v16bf a[MTS];
#pragma unroll
    for (int mt = 0; mt < MTS; ++mt) a[mt] = load_a_row16(hsh, mt, kt, lane);
#pragma unroll
    for (int ntl = 0; ntl < 2; ++ntl) {
      v16bf bf = load_b_frag(wt, kt * 16 + w * 2 + ntl, lane);
#pragma unroll
      for (int mt = 0; mt < MTS; ++mt)
        acc[ntl][mt] = wmma_bf16(a[mt], bf, acc[ntl][mt]);
    }
  }
  const int hi = lane >> 4;
#pragma unroll
  for (int ntl = 0; ntl < 2; ++ntl) {
    const int n = (w * 2 + ntl) * 16 + (lane & 15);
    const float bb = bias[n];
#pragma unroll
    for (int mt = 0; mt < MTS; ++mt)
#pragma unroll
      for (int v = 0; v < 8; ++v)
        dst[mt * 16 + v + 8 * hi][n] = acc[ntl][mt][v] + bb;
  }
}

// Encoder GEMM: hs_raw = concat(x[m],x[m+1]) @ W_in^T + b_in (K=512).
template <int MTS>
__device__ __forceinline__ void gemm_enc(const unsigned short* __restrict__ xbh,
                                         const unsigned short* __restrict__ wt_in,
                                         const float* __restrict__ b_in,
                                         float (*__restrict__ hs)[DD],
                                         int w, int lane) {
  v8f acc[2][MTS];
#pragma unroll
  for (int i = 0; i < 2; ++i)
#pragma unroll
    for (int j = 0; j < MTS; ++j) acc[i][j] = (v8f)0.0f;

#pragma unroll 2
  for (int kt = 0; kt < 16; ++kt) {
    v16bf a[MTS];
#pragma unroll
    for (int mt = 0; mt < MTS; ++mt) a[mt] = load_a_cat16(xbh, mt, kt, lane);
#pragma unroll
    for (int ntl = 0; ntl < 2; ++ntl) {
      v16bf bf = load_b_frag(wt_in, kt * 16 + w * 2 + ntl, lane);
#pragma unroll
      for (int mt = 0; mt < MTS; ++mt)
        acc[ntl][mt] = wmma_bf16(a[mt], bf, acc[ntl][mt]);
    }
  }
  const int hi = lane >> 4;
#pragma unroll
  for (int ntl = 0; ntl < 2; ++ntl) {
    const int n = (w * 2 + ntl) * 16 + (lane & 15);
    const float bb = b_in[n];
#pragma unroll
    for (int mt = 0; mt < MTS; ++mt)
#pragma unroll
      for (int v = 0; v < 8; ++v)
        hs[mt * 16 + v + 8 * hi][n] = acc[ntl][mt][v] + bb;
  }
}

// One full merge step, specialized on M-tile count (branch-free WMMA loops).
template <int MTS>
__device__ __forceinline__ void rae_step(
    int l, int nrows,
    float (*__restrict__ xb)[DD], float (*__restrict__ hs)[DD],
    float (*__restrict__ rec)[DD],
    unsigned short* __restrict__ xbh, unsigned short* __restrict__ hsh,
    float* __restrict__ perp, int* __restrict__ s_ind,
    const float* __restrict__ b_in, const float* __restrict__ b_outl,
    const float* __restrict__ b_outr,
    const unsigned short* __restrict__ wt_in,
    const unsigned short* __restrict__ wt_outl,
    const unsigned short* __restrict__ wt_outr,
    int w, int lane, int tid, float& accPerp) {
  // ---- encoder
  gemm_enc<MTS>(xbh, wt_in, b_in, hs, w, lane);
  __syncthreads();
  softmax4(hs, hsh, nrows, w, lane);       // hs = vecm rows (+bf16 mirror)
  __syncthreads();

  // ---- left decoder + KL(recl || vecl)
  gemm_dec<MTS>(hsh, wt_outl, b_outl, rec, w, lane);
  __syncthreads();
  softmax4(rec, nullptr, nrows, w, lane);
  __syncthreads();
#pragma unroll
  for (int rr = 0; rr < 4; ++rr) {
    const int r = w * 4 + rr;              // uniform per wave
    if (r < nrows) {
      const float kl = kl_row(&rec[r][0], &xb[r][0], lane);
      if (lane == 0) perp[r] = kl;
    }
  }
  __syncthreads();

  // ---- right decoder + KL(recr || vecr)
  gemm_dec<MTS>(hsh, wt_outr, b_outr, rec, w, lane);
  __syncthreads();
  softmax4(rec, nullptr, nrows, w, lane);
  __syncthreads();
#pragma unroll
  for (int rr = 0; rr < 4; ++rr) {
    const int r = w * 4 + rr;
    if (r < nrows) {
      const float kl = kl_row(&rec[r][0], &xb[r + 1][0], lane);
      if (lane == 0) perp[r] += kl;
    }
  }
  __syncthreads();

  // ---- argmax over pair perplexities (wave 0), first-index tiebreak
  if (w == 0) {
    float val = (lane < nrows) ? perp[lane] : -3.402823466e38f;
    int idx   = (lane < nrows) ? lane : LL;
#pragma unroll
    for (int o = 16; o; o >>= 1) {
      const float ov = __shfl_xor(val, o, 32);
      const int   oi = __shfl_xor(idx, o, 32);
      if (ov > val || (ov == val && oi < idx)) { val = ov; idx = oi; }
    }
    if (lane == 0) { *s_ind = idx; accPerp += val; }
  }
  __syncthreads();
  const int ind = *s_ind;

  // ---- merge: x[ind] = vecm, shift rows above down by one (both copies).
  // Each thread owns exactly one column -> race-free ascending shift.
  {
    const int e = tid;                     // 0..255
    xb[ind][e] = hs[ind][e];
    xbh[ind * BFP + e] = hsh[ind * BFP + e];
    for (int j = ind + 1; j <= l - 2; ++j) {
      xb[j][e] = xb[j + 1][e];
      xbh[j * BFP + e] = xbh[(j + 1) * BFP + e];
    }
  }
  __syncthreads();
}

// ---------------------------------------------------------------------------
// Weight pre-swizzle: W[n][k] fp32 -> bf16 in WMMA B-fragment order.
// Frag (kt,nt): lane L, vgpr v, half h -> W[nt*16 + L%16][kt*32+koff(v,L/16)+h]
// ---------------------------------------------------------------------------
__global__ void convert_weights(const float* __restrict__ W_in,
                                const float* __restrict__ W_outl,
                                const float* __restrict__ W_outr,
                                unsigned short* __restrict__ wt,
                                float* __restrict__ out) {
  const size_t gid = (size_t)blockIdx.x * 256 + threadIdx.x;
  if (gid == 0) out[0] = 0.0f;            // zero the scalar accumulator
  if (gid >= 262144) return;              // 131072 + 65536 + 65536 halves

  const float* W;
  int Kdim;
  size_t local;
  if (gid < 131072)       { W = W_in;   Kdim = 512; local = gid; }
  else if (gid < 196608)  { W = W_outl; Kdim = 256; local = gid - 131072; }
  else                    { W = W_outr; Kdim = 256; local = gid - 196608; }

  const int fi   = (int)(local / 512);
  const int rem  = (int)(local % 512);
  const int lane = rem / 16;
  const int h16  = rem % 16;
  const int v    = h16 >> 1;
  const int hh   = h16 & 1;
  const int kt   = fi / 16;
  const int nt   = fi % 16;
  const int hi   = lane >> 4;
  const int n    = nt * 16 + (lane & 15);
  const int k    = kt * 32 + koff(v, hi) + hh;

  wt[gid] = f2bf(W[(size_t)n * Kdim + k]);
}

// ---------------------------------------------------------------------------
// Persistent per-batch-column kernel: full 31-step recursion in LDS.
// ---------------------------------------------------------------------------
__global__ __launch_bounds__(256) void rae_kernel(
    const float* __restrict__ inp,
    const float* __restrict__ b_in,
    const float* __restrict__ b_outl,
    const float* __restrict__ b_outr,
    const unsigned short* __restrict__ wt_in,
    const unsigned short* __restrict__ wt_outl,
    const unsigned short* __restrict__ wt_outr,
    float* __restrict__ out) {
  __shared__ float xb[LL][DD];              // sequence (fp32, for KL q-terms)
  __shared__ float hs[LL][DD];              // hidden raw -> softmaxed (vecm)
  __shared__ float rec[LL][DD];             // reconstruction raw -> softmaxed
  __shared__ unsigned short xbh[LL * BFP];  // bf16 mirror of xb (padded rows)
  __shared__ unsigned short hsh[LL * BFP];  // bf16 mirror of softmaxed hs
  __shared__ float perp[LL];
  __shared__ int   s_ind;

  const int tid  = threadIdx.x;
  const int lane = tid & 31;
  const int w    = tid >> 5;       // 8 waves
  const int b    = blockIdx.x;     // batch column

  // load this batch column into LDS (fp32 + bf16 mirror)
  for (int i = tid; i < LL * DD; i += 256) {
    const int l = i >> 8, d = i & 255;
    const float f = inp[(size_t)l * BB * DD + (size_t)b * DD + d];
    xb[l][d] = f;
    xbh[l * BFP + d] = f2bf(f);
  }
  __syncthreads();

  float accPerp = 0.0f;            // only tid 0's copy is used

  // first 15 merges have >16 pair-rows -> 2 M-tiles; remaining 16 -> 1 tile.
  for (int it = 0; it < 15; ++it) {
    const int l = LL - it;
    rae_step<2>(l, l - 1, xb, hs, rec, xbh, hsh, perp, &s_ind,
                b_in, b_outl, b_outr, wt_in, wt_outl, wt_outr,
                w, lane, tid, accPerp);
  }
  for (int it = 15; it < LL - 1; ++it) {
    const int l = LL - it;
    rae_step<1>(l, l - 1, xb, hs, rec, xbh, hsh, perp, &s_ind,
                b_in, b_outl, b_outr, wt_in, wt_outl, wt_outr,
                w, lane, tid, accPerp);
  }

  if (tid == 0)
    atomicAdd(out, accPerp * (0.5f / ((float)BB * (float)LL)));
}

// ---------------------------------------------------------------------------
extern "C" void kernel_launch(void* const* d_in, const int* in_sizes, int n_in,
                              void* d_out, int out_size, void* d_ws, size_t ws_size,
                              hipStream_t stream) {
  const float* inp    = (const float*)d_in[0];
  const float* W_in   = (const float*)d_in[1];
  const float* b_in   = (const float*)d_in[2];
  const float* W_outl = (const float*)d_in[3];
  const float* b_outl = (const float*)d_in[4];
  const float* W_outr = (const float*)d_in[5];
  const float* b_outr = (const float*)d_in[6];
  float* out = (float*)d_out;

  unsigned short* wt = (unsigned short*)d_ws;   // 512 KB of bf16 fragments
  // zero out + swizzle weights into WMMA B-fragment order (bf16)
  convert_weights<<<1024, 256, 0, stream>>>(W_in, W_outl, W_outr, wt, out);
  // one persistent workgroup per batch column
  rae_kernel<<<BB, 256, 0, stream>>>(inp, b_in, b_outl, b_outr,
                                     wt,                 // W_in  frags (16x16)
                                     wt + 131072,        // W_outl frags (8x16)
                                     wt + 196608,        // W_outr frags (8x16)
                                     out);
}
